// LSTM_40742059769959
// MI455X (gfx1250) — matrix operand order repaired and checked by reading, not compiled, not run on hardware
//
#include <hip/hip_runtime.h>
#include <hip/hip_bf16.h>

#define S_LEN 512
#define B_SZ  64
#define IN_SZ 1024
#define H_SZ  1024
#define G_SZ  4096   // 4*H

typedef __attribute__((ext_vector_type(16))) __bf16 v16bf;
typedef __attribute__((ext_vector_type(8)))  __bf16 v8bf;
typedef __attribute__((ext_vector_type(8)))  float  v8f;
typedef __attribute__((ext_vector_type(4)))  int    v4i;

#ifndef __has_builtin
#define __has_builtin(x) 0
#endif

#if defined(__AMDGCN__) && __has_builtin(__builtin_amdgcn_global_load_async_to_lds_b128)
#define ASYNC_LDS 1
#else
#define ASYNC_LDS 0
#endif

// pointer-to-AS1-int4 / pointer-to-AS3-int4 (per builtin signature vV4i*1 V4i*3 Ii Ii)
typedef __attribute__((address_space(1))) v4i* v4i_gp;
typedef __attribute__((address_space(3))) v4i* v4i_lp;

// async 16B global->LDS copy (CDNA5 path), synchronous fallback otherwise
__device__ __forceinline__ void copy16_to_lds(const void* g, void* l) {
#if ASYNC_LDS
  __builtin_amdgcn_global_load_async_to_lds_b128((v4i_gp)g, (v4i_lp)l, 0, 0);
#else
  *(uint4*)l = *(const uint4*)g;
#endif
}
__device__ __forceinline__ void async_wait0() {
#if ASYNC_LDS
#if __has_builtin(__builtin_amdgcn_s_wait_asynccnt)
  __builtin_amdgcn_s_wait_asynccnt(0);
#else
  asm volatile("s_wait_asynccnt 0x0" ::: "memory");
#endif
#endif
}

// ---- WMMA operand loaders (contiguous k order for A and B, wave32) ----
// A 16x32 bf16: lane L holds row M=L&15. Lanes 0-15: K[0..15];
// lanes 16-31: K[16..31]. Caller passes p = row + k0 + 16*(lane>>4).
__device__ __forceinline__ v16bf load_A16x32(const __bf16* p) {
  v8bf lo = *reinterpret_cast<const v8bf*>(p);
  v8bf hi = *reinterpret_cast<const v8bf*>(p + 8);
  return __builtin_shufflevector(lo, hi, 0,1,2,3,4,5,6,7,8,9,10,11,12,13,14,15);
}
// B 32x16 bf16: lane L holds column N=L&15; 16 contiguous K per half-wave.
__device__ __forceinline__ v16bf load_B32x16(const void* p) {
  return *reinterpret_cast<const v16bf*>(p);   // 32B aligned
}

__device__ __forceinline__ float sigmoid_(float x) {
  return 1.0f / (1.0f + __expf(-x));
}
__device__ __forceinline__ float tanh_(float x) {
  float e = __expf(-2.0f * fabsf(x));
  float r = (1.0f - e) / (1.0f + e);
  return copysignf(r, x);
}

// ---- small utility kernels ----
__global__ void cvt_f32_bf16_k(const float* __restrict__ s, __bf16* __restrict__ d, int n) {
  for (int i = blockIdx.x * blockDim.x + threadIdx.x; i < n; i += gridDim.x * blockDim.x)
    d[i] = (__bf16)s[i];
}
__global__ void bias_sum_k(const float* __restrict__ a, const float* __restrict__ b,
                           float* __restrict__ o, int n) {
  for (int i = blockIdx.x * blockDim.x + threadIdx.x; i < n; i += gridDim.x * blockDim.x)
    o[i] = a[i] + b[i];
}
__global__ void zero_state_k(__bf16* __restrict__ h, float* __restrict__ c, int n) {
  for (int i = blockIdx.x * blockDim.x + threadIdx.x; i < n; i += gridDim.x * blockDim.x) {
    h[i] = (__bf16)0.0f;
    c[i] = 0.0f;
  }
}
__global__ void copy_f32_k(const float* __restrict__ s, float* __restrict__ d, int n) {
  for (int i = blockIdx.x * blockDim.x + threadIdx.x; i < n; i += gridDim.x * blockDim.x)
    d[i] = s[i];
}

// ---- big non-recurrent GEMM: out[M,4096] = A[M,1024] @ W[4096,1024]^T + bias ----
// block = 128 threads (4 waves); block tile 64 rows x 64 cols; wave tile 16x64.
// B panel (64 contiguous W rows = 128KB bf16) async-staged to LDS once per
// block and shared by the 4 waves; K-loop reads B via ds_load, A from global.
__global__ __launch_bounds__(128, 2)
void pregates_gemm(const __bf16* __restrict__ A, const __bf16* __restrict__ W,
                   const float* __restrict__ bias, float* __restrict__ out) {
  extern __shared__ char smem[];       // 131072 B: W rows nBase..nBase+63
  const int tid   = threadIdx.x;
  const int lane  = tid & 31;
  const int wave  = tid >> 5;
  const int mBase = blockIdx.x * 64 + wave * 16;
  const int nBase = blockIdx.y * 64;
  const int l16   = lane & 15;
  const int half  = lane >> 4;

  // stage B panel: one contiguous 128KB range of W
  const char* wPanel = (const char*)W + (size_t)nBase * IN_SZ * 2;
  for (int q = tid; q < 8192; q += 128)
    copy16_to_lds(wPanel + (size_t)q * 16, smem + (size_t)q * 16);
  async_wait0();
  __syncthreads();

  const __bf16* aBase = A + (size_t)(mBase + l16) * IN_SZ + (half << 4);
  const char*   bBase = smem + ((size_t)l16 * IN_SZ + (half << 4)) * 2;

  v8f acc[4] = {};
#pragma unroll
  for (int k0 = 0; k0 < IN_SZ; k0 += 32) {
    v16bf a = load_A16x32(aBase + k0);
#pragma unroll
    for (int g = 0; g < 4; ++g) {
      v16bf b = load_B32x16(bBase + (size_t)g * 32768 + k0 * 2);
      acc[g] = __builtin_amdgcn_wmma_f32_16x16x32_bf16(false, a, false, b,
                                                       (short)0, acc[g], false, false);
    }
  }
  // C/D layout: lane holds N=lane&15; VGPR r -> M = r + 8*(lane>>4)
  const int mOff = mBase + (half << 3);
#pragma unroll
  for (int g = 0; g < 4; ++g) {
    const int n = nBase + g * 16 + l16;
    const float bv = bias[n];
#pragma unroll
    for (int r = 0; r < 8; ++r)
      out[(size_t)(mOff + r) * G_SZ + n] = acc[g][r] + bv;
  }
}

// ---- one recurrent time step, fully fused ----
// 64 blocks x 256 threads (8 waves). Block owns hidden cols [j0,j0+16) for all
// four gates. W_hh panel (4 x 16 rows x 1024 K = 128KB bf16) is async-staged
// into LDS once, shared by all waves. Waves 0-3 take K[0,512), waves 4-7 take
// K[512,1024); partials reduced through LDS; waves 0-3 run the fused epilogue.
__global__ __launch_bounds__(256, 2)
void lstm_step(const __bf16* __restrict__ Whh,      // [4096,1024] bf16
               const float*  __restrict__ pre_t,    // [64,4096] x@Wih^T + bias
               const __bf16* __restrict__ h_in,     // [64,1024] (t-1)
               __bf16*       __restrict__ h_out,    // [64,1024] (t)  ping-pong
               float*        __restrict__ c_state,  // [64,1024] in/out
               float*        __restrict__ h_f32,    // [64,1024] fp32 state out
               __bf16*       __restrict__ hseq,     // layer0: bf16 seq out, else null
               float*        __restrict__ oseq) {   // layer1: d_out seq, else null
  extern __shared__ char smem[];
  char* bpanel = smem;                 // 131072 B: 4 gate chunks of 32KB
  char* redbuf = smem + 131072;        // 16384 B: cross-wave K reduction

  const int tid  = threadIdx.x;
  const int lane = tid & 31;
  const int wave = tid >> 5;
  const int mq   = wave & 3;           // M quarter (rows 16*mq..)
  const int kh   = wave >> 2;          // K half
  const int j0   = blockIdx.x * 16;
  const int l16  = lane & 15;
  const int half = lane >> 4;

  // ---- stage B panel: rows (g*1024 + j0 .. +15) x K=1024, 4 contiguous 32KB chunks
  for (int q = tid; q < 8192; q += 256) {
    const int g = q >> 11;             // 2048 x 16B per gate chunk
    const int r = q & 2047;
    const char* gp = (const char*)Whh + ((size_t)(g * 1024 + j0) * H_SZ) * 2 + (size_t)r * 16;
    copy16_to_lds(gp, bpanel + (size_t)g * 32768 + (size_t)r * 16);
  }
  async_wait0();
  __syncthreads();

  const int kbeg = kh * 512;
  const __bf16* aBase = h_in + (size_t)(mq * 16 + l16) * H_SZ + (half << 4) + kbeg;
  const char*   bBase = bpanel + ((size_t)l16 * H_SZ + (half << 4) + kbeg) * 2;

  v8f acc[4] = {};
#pragma unroll
  for (int kk = 0; kk < 512; kk += 32) {
    v16bf a = load_A16x32(aBase + kk);
#pragma unroll
    for (int g = 0; g < 4; ++g) {
      v16bf b = load_B32x16(bBase + (size_t)g * 32768 + kk * 2);
      acc[g] = __builtin_amdgcn_wmma_f32_16x16x32_bf16(false, a, false, b,
                                                       (short)0, acc[g], false, false);
    }
  }

  // ---- reduce the two K halves through LDS ----
  __syncthreads();                     // all B panel reads complete
  if (kh == 1) {
#pragma unroll
    for (int g = 0; g < 4; ++g)
      *(v8f*)(redbuf + ((size_t)(mq * 4 + g) * 1024) + (size_t)lane * 32) = acc[g];
  }
  __syncthreads();
  if (kh == 1) return;
#pragma unroll
  for (int g = 0; g < 4; ++g) {
    v8f o = *(const v8f*)(redbuf + ((size_t)(mq * 4 + g) * 1024) + (size_t)lane * 32);
#pragma unroll
    for (int r = 0; r < 8; ++r) acc[g][r] += o[r];
  }

  // ---- fused LSTM cell epilogue (waves 0-3 cover all 64 batch rows) ----
  const int n = j0 + l16;
  const int bBat = mq * 16 + (half << 3);
#pragma unroll
  for (int r = 0; r < 8; ++r) {
    const int b = bBat + r;
    const size_t pb = (size_t)b * G_SZ + n;
    const float iv = sigmoid_(acc[0][r] + pre_t[pb]);
    const float fv = sigmoid_(acc[1][r] + pre_t[pb + 1024]);
    const float gv = tanh_  (acc[2][r] + pre_t[pb + 2048]);
    const float ov = sigmoid_(acc[3][r] + pre_t[pb + 3072]);
    const size_t hb = (size_t)b * H_SZ + n;
    const float cn = fv * c_state[hb] + iv * gv;
    const float hv = ov * tanh_(cn);
    c_state[hb] = cn;
    h_f32[hb]   = hv;
    const __bf16 hv16 = (__bf16)hv;
    h_out[hb] = hv16;
    if (hseq) hseq[hb] = hv16;
    if (oseq) oseq[hb] = hv;
  }
}

extern "C" void kernel_launch(void* const* d_in, const int* in_sizes, int n_in,
                              void* d_out, int out_size, void* d_ws, size_t ws_size,
                              hipStream_t stream) {
  (void)in_sizes; (void)n_in; (void)out_size; (void)ws_size;
  const float* x   = (const float*)d_in[0];
  const float* wih = (const float*)d_in[1];
  const float* whh = (const float*)d_in[2];
  const float* bih = (const float*)d_in[3];
  const float* bhh = (const float*)d_in[4];
  float* out = (float*)d_out;

  char* ws = (char*)d_ws;
  size_t off = 0;
  auto carve = [&](size_t bytes) -> char* {
    char* p = ws + off;
    off += (bytes + 255) & ~(size_t)255;
    return p;
  };
  const size_t M = (size_t)S_LEN * B_SZ;                       // 32768 rows
  float*  pregates = (float*) carve(M * G_SZ * sizeof(float)); // 512 MB, reused per layer
  __bf16* x_bf     = (__bf16*)carve(M * IN_SZ * sizeof(__bf16));
  __bf16* l0_bf    = (__bf16*)carve(M * H_SZ * sizeof(__bf16));
  __bf16* wih_bf   = (__bf16*)carve((size_t)G_SZ * IN_SZ * sizeof(__bf16));
  __bf16* whh_bf   = (__bf16*)carve((size_t)G_SZ * H_SZ * sizeof(__bf16));
  float*  bias     = (float*) carve(G_SZ * sizeof(float));
  __bf16* hping0   = (__bf16*)carve((size_t)B_SZ * H_SZ * sizeof(__bf16));
  __bf16* hping1   = (__bf16*)carve((size_t)B_SZ * H_SZ * sizeof(__bf16));
  float*  c_state  = (float*) carve((size_t)B_SZ * H_SZ * sizeof(float));
  float*  h_f32    = (float*) carve((size_t)B_SZ * H_SZ * sizeof(float));

  cvt_f32_bf16_k<<<4096, 256, 0, stream>>>(x, x_bf, (int)(M * IN_SZ));
  cvt_f32_bf16_k<<<2048, 256, 0, stream>>>(wih, wih_bf, G_SZ * IN_SZ);
  cvt_f32_bf16_k<<<2048, 256, 0, stream>>>(whh, whh_bf, G_SZ * H_SZ);
  bias_sum_k<<<16, 256, 0, stream>>>(bih, bhh, bias, G_SZ);

  float* outputs = out;                                  // [S,B,H]
  float* h_n = out + M * H_SZ;                           // [2,B,H]
  float* c_n = h_n + 2 * (size_t)B_SZ * H_SZ;            // [2,B,H]
  __bf16* hp[2] = {hping0, hping1};
  const size_t shmemStep = 131072 + 16384;
  const size_t shmemGemm = 131072;

  for (int layer = 0; layer < 2; ++layer) {
    const __bf16* inp = (layer == 0) ? x_bf : l0_bf;
    dim3 gg((unsigned)(M / 64), G_SZ / 64);
    pregates_gemm<<<gg, 128, shmemGemm, stream>>>(inp, wih_bf, bias, pregates);
    zero_state_k<<<64, 256, 0, stream>>>(hping0, c_state, B_SZ * H_SZ);
    for (int t = 0; t < S_LEN; ++t) {
      lstm_step<<<64, 256, shmemStep, stream>>>(
          whh_bf, pregates + (size_t)t * B_SZ * G_SZ,
          hp[t & 1], hp[(t + 1) & 1], c_state, h_f32,
          (layer == 0) ? (l0_bf + (size_t)t * B_SZ * H_SZ) : (__bf16*)nullptr,
          (layer == 1) ? (outputs + (size_t)t * B_SZ * H_SZ) : (float*)nullptr);
    }
    copy_f32_k<<<64, 256, 0, stream>>>(h_f32, h_n + (size_t)layer * B_SZ * H_SZ, B_SZ * H_SZ);
    copy_f32_k<<<64, 256, 0, stream>>>(c_state, c_n + (size_t)layer * B_SZ * H_SZ, B_SZ * H_SZ);
  }
}
